// EncoderRNN_43310450213554
// MI455X (gfx1250) — matrix operand order repaired and checked by reading, not compiled, not run on hardware
//
#include <hip/hip_runtime.h>

#define H    1024
#define SEQ  4096
#define G3   3072
#define HP   1028          // H + 4
#define NWG_SCAN 64
#define SCAN_T   384       // 48 dots x 8 lanes
#define CH   16            // h outputs owned per scan block

#define BM 128
#define BN 64
#define BK 32

typedef __attribute__((ext_vector_type(16))) __bf16 v16bf;
typedef __attribute__((ext_vector_type(8)))  float  v8f;

__device__ __forceinline__ unsigned short f32_to_bf16_bits(float f) {
  unsigned u = __builtin_bit_cast(unsigned, f);
  u += 0x7FFFu + ((u >> 16) & 1u);               // round-to-nearest-even
  return (unsigned short)(u >> 16);
}
__device__ __forceinline__ __bf16 bits_to_bf16(unsigned short s) {
  return __builtin_bit_cast(__bf16, s);
}
__device__ __forceinline__ float sigmoidf_(float x) {
  return 1.0f / (1.0f + __expf(-x));
}

// CDNA5 async memory->LDS copy (16B per lane), tracked by ASYNCcnt.
__device__ __forceinline__ void async_copy16(unsigned lds_off, const void* gptr) {
#ifdef __HIP_DEVICE_COMPILE__
  asm volatile("global_load_async_to_lds_b128 %0, %1, off"
               :: "v"(lds_off), "v"((unsigned long long)(__SIZE_TYPE__)gptr)
               : "memory");
#endif
}
__device__ __forceinline__ void wait_async0() {
#ifdef __HIP_DEVICE_COMPILE__
  asm volatile("s_wait_asynccnt 0x0" ::: "memory");
#endif
}

// ---------------- kernel 1: h0 = W_reformat @ hidden + b ; init barrier ----
__global__ __launch_bounds__(256) void init_h_kernel(
    const float* __restrict__ hidden, const float* __restrict__ Wr,
    const float* __restrict__ br, float* __restrict__ hbuf,
    unsigned* __restrict__ bar) {
  int j = blockIdx.x * blockDim.x + threadIdx.x;
  if (j == 0) { bar[0] = 0u; bar[1] = 0u; }
  if (j < H) {
    const float* w = Wr + (size_t)j * HP;
    float acc = 0.f;
    #pragma unroll 4
    for (int k = 0; k < HP; ++k) acc += w[k] * hidden[k];
    hbuf[j] = acc + br[j];                       // ping-pong buffer 0
  }
}

// ---------------- kernel 2: gather embeddings -> bf16 A ------------------
__global__ __launch_bounds__(256) void gather_cast_A(
    const int* __restrict__ tokens, const float* __restrict__ emb,
    unsigned short* __restrict__ Abf) {
  size_t i = (size_t)blockIdx.x * blockDim.x + threadIdx.x;  // SEQ*H
  if (i < (size_t)SEQ * H) {
    int s = (int)(i >> 10);
    int k = (int)(i & (H - 1));
    Abf[i] = f32_to_bf16_bits(emb[(size_t)tokens[s] * H + k]);
  }
}

// ---------------- kernel 3: W_ih -> bf16 B -------------------------------
__global__ __launch_bounds__(256) void cast_B(
    const float* __restrict__ Wih, unsigned short* __restrict__ Bbf) {
  size_t i = (size_t)blockIdx.x * blockDim.x + threadIdx.x;
  if (i < (size_t)G3 * H) Bbf[i] = f32_to_bf16_bits(Wih[i]);
}

// ---------------- kernel 4: x_gates = A @ B^T + b_ih  (WMMA bf16) --------
// Block = 128 threads (4 waves), block tile 128x64, K-step 32.
// Wave w owns rows [w*32, w*32+32): 2 A-frags x 4 B-frags = 8 WMMA / K-step.
// LDS tiles are double-buffered and filled with GLOBAL_LOAD_ASYNC_TO_LDS_B128
// so global fetch of slab k+1 overlaps WMMA on slab k.
__global__ __launch_bounds__(128) void gemm_xg_kernel(
    const unsigned short* __restrict__ Abf,   // [SEQ][H]  (bf16 bits)
    const unsigned short* __restrict__ Bbf,   // [G3][H]   (bf16 bits, N-major K-contig)
    const float* __restrict__ b_ih,
    float* __restrict__ XG) {                 // [SEQ][G3] fp32
  __shared__ __align__(16) unsigned short Asub[2][BM][BK];   // 16 KB
  __shared__ __align__(16) unsigned short Bsub[2][BN][BK];   //  8 KB
  const int tid  = threadIdx.x;
  const int wid  = tid >> 5;
  const int lane = tid & 31;
  const int m_tile = blockIdx.y * BM;
  const int n_tile = blockIdx.x * BN;

  v8f acc[2][4];
  for (int h2 = 0; h2 < 2; ++h2)
    for (int n = 0; n < 4; ++n)
      for (int r = 0; r < 8; ++r) acc[h2][n][r] = 0.0f;

  // issue async stage of one K-slab into buffer bsel
  auto stage = [&](int bsel, int k0) {
    #pragma unroll
    for (int i = 0; i < 4; ++i) {              // A: 512 chunks of 8 bf16
      int idx = tid + i * 128;
      int row = idx >> 2, ch = idx & 3;
      async_copy16((unsigned)(__SIZE_TYPE__)&Asub[bsel][row][ch * 8],
                   Abf + (size_t)(m_tile + row) * H + k0 + ch * 8);
    }
    #pragma unroll
    for (int i = 0; i < 2; ++i) {              // B: 256 chunks
      int idx = tid + i * 128;
      int row = idx >> 2, ch = idx & 3;
      async_copy16((unsigned)(__SIZE_TYPE__)&Bsub[bsel][row][ch * 8],
                   Bbf + (size_t)(n_tile + row) * H + k0 + ch * 8);
    }
  };

  stage(0, 0);
  wait_async0();
  __syncthreads();

  const int abase = (lane < 16) ? 0 : 16;      // A-frag K layout (contiguous, matches B)
  const int ncol  = lane & 15;
  const int kbase = (lane < 16) ? 0 : 16;      // B-frag K layout

  for (int it = 0; it < H / BK; ++it) {
    const int cur = it & 1;
    if (it + 1 < H / BK) stage(cur ^ 1, (it + 1) * BK);   // async prefetch next slab

    v16bf af[2];
    #pragma unroll
    for (int h2 = 0; h2 < 2; ++h2) {
      const int m = wid * 32 + h2 * 16 + (lane & 15);
      #pragma unroll
      for (int j = 0; j < 16; ++j) {
        int kk = abase + j;
        af[h2][j] = bits_to_bf16(Asub[cur][m][kk]);
      }
    }
    v16bf bfr[4];
    #pragma unroll
    for (int n = 0; n < 4; ++n)
      #pragma unroll
      for (int j = 0; j < 16; ++j)
        bfr[n][j] = bits_to_bf16(Bsub[cur][n * 16 + ncol][kbase + j]);

#ifdef __HIP_DEVICE_COMPILE__
    #pragma unroll
    for (int h2 = 0; h2 < 2; ++h2)
      #pragma unroll
      for (int n = 0; n < 4; ++n)
        acc[h2][n] = __builtin_amdgcn_wmma_f32_16x16x32_bf16(
            false, af[h2], false, bfr[n], (short)0, acc[h2][n], false, false);
#endif
    wait_async0();           // next slab landed in the other buffer
    __syncthreads();         // all waves done reading current slab
  }

  // D layout: VGPR r -> M=r (lanes 0-15) / M=8+r (lanes 16-31), N=lane&15
  #pragma unroll
  for (int h2 = 0; h2 < 2; ++h2) {
    const int mrow0 = m_tile + wid * 32 + h2 * 16 + ((lane < 16) ? 0 : 8);
    #pragma unroll
    for (int n = 0; n < 4; ++n) {
      int g = n_tile + n * 16 + (lane & 15);
      float bias = b_ih[g];
      #pragma unroll
      for (int r = 0; r < 8; ++r)
        XG[(size_t)(mrow0 + r) * G3 + g] = acc[h2][n][r] + bias;
    }
  }
}

// ---------------- kernel 5: persistent GRU scan --------------------------
// 64 blocks x 384 threads; block b owns h[b*16 .. b*16+16). Its 48 fixed
// W_hh rows live in LDS (192 KB of the 320 KB WGP LDS) for all 4096 steps.
__global__ __launch_bounds__(SCAN_T) void gru_scan_kernel(
    const float* __restrict__ Whh, const float* __restrict__ bhh,
    const float* __restrict__ XG, float* __restrict__ hbuf,
    unsigned* __restrict__ bar) {
  extern __shared__ float smem[];
  float* ldsW = smem;                 // 48*1024 floats
  float* ldsh = smem + 48 * H;        // 1024 floats
  __shared__ float gates[48];

  const int tid = threadIdx.x;
  const int blk = blockIdx.x;

  // one-time stage of this block's 48 W_hh rows
  for (int i = tid; i < 48 * H; i += SCAN_T) {
    int lr = i >> 10;
    int k  = i & (H - 1);
    int grow = (lr >> 4) * H + blk * CH + (lr & 15);
    ldsW[i] = Whh[(size_t)grow * H + k];
  }

  const int d = tid >> 3;             // dot index 0..47 (gate*16 + jl)
  const int c = tid & 7;              // k-chunk within dot
  const float bhh_d = bhh[(d >> 4) * H + blk * CH + (d & 15)];

  unsigned* cnt = bar;
  unsigned* phs = bar + 1;

  for (int t = 0; t < SEQ; ++t) {
    const float* hsrc = hbuf + (size_t)(t & 1) * H;
    float*       hdst = hbuf + (size_t)((t + 1) & 1) * H;
    for (int i = tid; i < H; i += SCAN_T) ldsh[i] = hsrc[i];
    if (tid < CH && t + 1 < SEQ)   // warm L2 for next step's x_gates slice
      __builtin_prefetch(XG + (size_t)(t + 1) * G3 + blk * CH + tid, 0, 1);
    __syncthreads();

    const float* wr = ldsW + d * H + c * 128;
    const float* hh = ldsh + c * 128;
    float p = 0.f;
    #pragma unroll 8
    for (int k = 0; k < 128; ++k) p += wr[k] * hh[k];
    p += __shfl_down(p, 4, 8);
    p += __shfl_down(p, 2, 8);
    p += __shfl_down(p, 1, 8);
    if (c == 0) gates[d] = p + bhh_d;
    __syncthreads();

    if (tid < CH) {
      int j = blk * CH + tid;
      const float* xg = XG + (size_t)t * G3;
      float rr = sigmoidf_(xg[j] + gates[tid]);
      float zz = sigmoidf_(xg[H + j] + gates[CH + tid]);
      float nn = tanhf(xg[2 * H + j] + rr * gates[2 * CH + tid]);
      hdst[j] = (1.f - zz) * nn + zz * ldsh[j];
      __threadfence();
    }

    // device-wide phase barrier (64 co-resident blocks)
    __syncthreads();
    if (tid == 0) {
      unsigned old = __hip_atomic_load(phs, __ATOMIC_ACQUIRE, __HIP_MEMORY_SCOPE_AGENT);
      __threadfence();
      unsigned tk = __hip_atomic_fetch_add(cnt, 1u, __ATOMIC_ACQ_REL, __HIP_MEMORY_SCOPE_AGENT);
      if (tk == NWG_SCAN - 1) {
        __hip_atomic_store(cnt, 0u, __ATOMIC_RELAXED, __HIP_MEMORY_SCOPE_AGENT);
        __hip_atomic_fetch_add(phs, 1u, __ATOMIC_RELEASE, __HIP_MEMORY_SCOPE_AGENT);
      } else {
        while (__hip_atomic_load(phs, __ATOMIC_ACQUIRE, __HIP_MEMORY_SCOPE_AGENT) == old) {
#ifdef __HIP_DEVICE_COMPILE__
          __builtin_amdgcn_s_sleep(2);
#endif
        }
      }
    }
    __syncthreads();
  }
}

// ---------------- kernel 6: mean / logvar projections --------------------
__global__ __launch_bounds__(256) void final_proj_kernel(
    const float* __restrict__ h,
    const float* __restrict__ Wm, const float* __restrict__ bm,
    const float* __restrict__ Wl, const float* __restrict__ bl,
    float* __restrict__ out) {
  int o = blockIdx.x * blockDim.x + threadIdx.x;
  if (o >= 2 * H) return;
  const float* w = (o < H) ? (Wm + (size_t)o * H) : (Wl + (size_t)(o - H) * H);
  float b = (o < H) ? bm[o] : bl[o - H];
  float acc = 0.f;
  #pragma unroll 4
  for (int k = 0; k < H; ++k) acc += w[k] * h[k];
  out[o] = acc + b;
}

extern "C" void kernel_launch(void* const* d_in, const int* in_sizes, int n_in,
                              void* d_out, int out_size, void* d_ws, size_t ws_size,
                              hipStream_t stream) {
  (void)in_sizes; (void)n_in; (void)out_size; (void)ws_size;
  const int*   tokens = (const int*)d_in[0];
  const float* hidden = (const float*)d_in[1];
  const float* emb    = (const float*)d_in[2];
  const float* Wr     = (const float*)d_in[3];
  const float* br     = (const float*)d_in[4];
  const float* Wih    = (const float*)d_in[5];
  const float* bih    = (const float*)d_in[6];
  const float* Whh    = (const float*)d_in[7];
  const float* bhh    = (const float*)d_in[8];
  const float* Wm     = (const float*)d_in[9];
  const float* bm     = (const float*)d_in[10];
  const float* Wl     = (const float*)d_in[11];
  const float* bl     = (const float*)d_in[12];
  float* out = (float*)d_out;

  char* ws = (char*)d_ws;
  float*          hbuf = (float*)ws;                              // 2*H floats
  unsigned*       bar  = (unsigned*)(ws + 8192);                  // 2 uints
  unsigned short* Abf  = (unsigned short*)(ws + 16384);           // SEQ*H bf16
  unsigned short* Bbf  = (unsigned short*)(ws + 16384 + (size_t)SEQ * H * 2);
  float*          XG   = (float*)(ws + 16384 + (size_t)SEQ * H * 2
                                          + (size_t)G3 * H * 2);  // SEQ*G3 fp32

  init_h_kernel<<<4, 256, 0, stream>>>(hidden, Wr, br, hbuf, bar);
  gather_cast_A<<<(SEQ * H) / 256, 256, 0, stream>>>(tokens, emb, Abf);
  cast_B<<<(G3 * H) / 256, 256, 0, stream>>>(Wih, Bbf);
  gemm_xg_kernel<<<dim3(G3 / BN, SEQ / BM), 128, 0, stream>>>(Abf, Bbf, bih, XG);

  size_t scan_lds = (size_t)(48 * H + H) * sizeof(float);         // ~196.6 KB
  hipFuncSetAttribute(reinterpret_cast<const void*>(gru_scan_kernel),
                      hipFuncAttributeMaxDynamicSharedMemorySize, (int)scan_lds);
  gru_scan_kernel<<<NWG_SCAN, SCAN_T, scan_lds, stream>>>(Whh, bhh, XG, hbuf, bar);

  final_proj_kernel<<<8, 256, 0, stream>>>(hbuf, Wm, bm, Wl, bl, out);
}
